// ScaledDotProductAttentionLayer_38027640439118
// MI455X (gfx1250) — compile-verified, matrix operand order checked by reading, not run on hardware
//
#include <hip/hip_runtime.h>
#include <math.h>

typedef __attribute__((ext_vector_type(16))) _Float16 v16h;
typedef __attribute__((ext_vector_type(8)))  _Float16 v8h;
typedef __attribute__((ext_vector_type(4)))  _Float16 v4h;
typedef __attribute__((ext_vector_type(8)))  float    v8f;
typedef __attribute__((ext_vector_type(4)))  unsigned int u32x4;
typedef __attribute__((ext_vector_type(8)))  int      i32x8;
typedef __attribute__((ext_vector_type(4)))  int      i32x4;

#define BHN   64
#define SEQ   1024
#define DK    64
#define BN    32      // keys per iteration
#define WAVES 4       // waves per block (wave32)
#define BM    (16*WAVES)
#define SKF   68      // K tile f32 row stride in floats (TDM pads 4 DW per 64-DW row -> 272B pitch)
#define SVST  40      // LDS V^T row stride in halfs (80B: 16B-aligned rows)

// Load 8 consecutive f32 -> elements [base..base+7] of a v16h (A/B operand half)
__device__ inline void load8_f16(const float* p, v16h& dst, int base) {
    float4 a = ((const float4*)p)[0];
    float4 b = ((const float4*)p)[1];
    dst[base+0] = (_Float16)a.x; dst[base+1] = (_Float16)a.y;
    dst[base+2] = (_Float16)a.z; dst[base+3] = (_Float16)a.w;
    dst[base+4] = (_Float16)b.x; dst[base+5] = (_Float16)b.y;
    dst[base+6] = (_Float16)b.z; dst[base+7] = (_Float16)b.w;
}

// Issue a TDM load of one K tile (BN x DK f32, row-major) into LDS with
// 4-DWORD padding after every 64-DWORD row (row pitch = SKF floats = 272B).
// D# layout per CDNA5 ISA ch.10.8 (group0: 128b, group1: 256b).
__device__ inline void tdm_load_K(const float* gsrc, unsigned lds_off) {
    unsigned long long ga = (unsigned long long)(uintptr_t)gsrc;
    u32x4 g0;
    g0[0] = 1u;                                     // count=1, user descriptor
    g0[1] = lds_off;                                // lds_addr (bytes)
    g0[2] = (unsigned)(ga & 0xFFFFFFFFull);         // global_addr[31:0]
    g0[3] = (unsigned)((ga >> 32) & 0x01FFFFFFull)  // global_addr[56:32]
          | (2u << 30);                             // type = 2 ("image")
    i32x8 g1;
    g1[0] = (2 << 16)       // data_size = 4 bytes
          | (1 << 20)       // pad_enable
          | (5 << 22)       // pad_interval code 5 -> every 64 DWORDs
          | (3 << 25);      // pad_amount  code 3 -> 4 DWORDs
    g1[1] = (DK << 16);     // tensor_dim0[15:0] = 64
    g1[2] = (BN << 16);     // tensor_dim1[15:0] = 32
    g1[3] = (DK << 16);     // tile_dim0 = 64
    g1[4] = BN;             // tile_dim1 = 32, tile_dim2 = 0
    g1[5] = DK;             // tensor_dim0_stride[31:0] = 64 elements
    g1[6] = 0;
    g1[7] = 0;
    i32x4 g2; g2[0]=0; g2[1]=0; g2[2]=0; g2[3]=0;   // 2D tile: groups 2/3 zero
    i32x4 g3; g3[0]=0; g3[1]=0; g3[2]=0; g3[3]=0;
    i32x8 gx; gx[0]=0; gx[1]=0; gx[2]=0; gx[3]=0;   // extra operand (clang-23 6-arg form)
    gx[4]=0; gx[5]=0; gx[6]=0; gx[7]=0;
    __builtin_amdgcn_tensor_load_to_lds(g0, g1, g2, g3, gx, 0);
}

__global__ void __launch_bounds__(128)
attn_fwd_kernel(const float* __restrict__ Q, const float* __restrict__ K,
                const float* __restrict__ V, const int* __restrict__ MSK,
                float* __restrict__ O)
{
    __shared__ float    ldsKf[2][BN * SKF];  // double-buffered K tile, f32 [key][d], TDM-filled
    __shared__ _Float16 ldsV[DK * SVST];     // V tile transposed, f16, [d][key]

    const int bh   = blockIdx.y;
    const int qblk = blockIdx.x;
    const int tid  = threadIdx.x;
    const int wave = tid >> 5;
    const int lane = tid & 31;
    const int ql   = lane & 15;   // query (N) index within wave tile
    const int hi   = lane >> 4;   // lane-half selector for K-dim swizzle

    const int qg = qblk * BM + wave * 16 + ql;            // global query row
    const float* qrow = Q   + ((size_t)bh * SEQ + qg) * DK;
    const int*   mrow = MSK + (size_t)qg * SEQ;
    const float* kbase = K + (size_t)bh * SEQ * DK;

    const unsigned koff[2] = { (unsigned)(uintptr_t)&ldsKf[0][0],
                               (unsigned)(uintptr_t)&ldsKf[1][0] };

    // ---- Q as B-operand (Q^T: N=q per lane, K=d swizzled), two 32-d chunks ----
    v16h qb[2];
    #pragma unroll
    for (int c = 0; c < 2; ++c) {
        load8_f16(qrow + 32*c +      hi*8, qb[c], 0);
        load8_f16(qrow + 32*c + 16 + hi*8, qb[c], 8);
    }

    // ---- flash-attention state (q = lane&15; both lane-halves keep full state) ----
    float m_run = -INFINITY;
    float l_run = 0.0f;
    v8f o[4];   // O^T accumulators: chunk c covers d = c*16 + (j + 8*hi)
    #pragma unroll
    for (int c = 0; c < 4; ++c)
        #pragma unroll
        for (int j = 0; j < 8; ++j) o[c][j] = 0.0f;

    const int krow = tid >> 2;    // staging: key row 0..31
    const int quad = tid & 3;     // staging: 4 threads per row

    // ---- prologue: DMA first K tile into buffer 0 ----
    if (tid < 32) tdm_load_K(kbase, koff[0]);

    const int NIT = SEQ / BN;
    for (int it = 0; it < NIT; ++it) {
        const int kt = it * BN;
        __syncthreads();   // previous iteration's readers (V + next K buffer) done

        // ---- stage V tile: global f32 [key][d] -> LDS f16 [d][key] (transpose) ----
        {
            const float* vr = V + ((size_t)bh * SEQ + kt + krow) * DK;
            #pragma unroll
            for (int i = 0; i < 4; ++i) {
                int f4 = quad + 4*i;
                float4 y = ((const float4*)vr)[f4];
                ldsV[(f4*4+0) * SVST + krow] = (_Float16)y.x;
                ldsV[(f4*4+1) * SVST + krow] = (_Float16)y.y;
                ldsV[(f4*4+2) * SVST + krow] = (_Float16)y.z;
                ldsV[(f4*4+3) * SVST + krow] = (_Float16)y.w;
            }
        }
        if (tid < 32) __builtin_amdgcn_s_wait_tensorcnt(0);  // current K tile arrived
        __syncthreads();   // K[cur] + V visible to all waves

        // ---- kick DMA for the NEXT K tile; overlaps with compute below ----
        if ((tid < 32) && (it + 1 < NIT))
            tdm_load_K(kbase + (size_t)(kt + BN) * DK, koff[(it + 1) & 1]);

        const float* kb = &ldsKf[it & 1][0];

        // ---- S^T = K_tile x Q^T : two 16-key sub-tiles, K-dim = 64 in 2 chunks ----
        v8f acc0 = {}; v8f acc1 = {};
        #pragma unroll
        for (int c = 0; c < 2; ++c) {
            v16h ka0, ka1;   // A operand: M=key row per lane, K=d swizzled
            load8_f16(kb + (     ql)*SKF + 32*c +      hi*8, ka0, 0);
            load8_f16(kb + (     ql)*SKF + 32*c + 16 + hi*8, ka0, 8);
            load8_f16(kb + (16 + ql)*SKF + 32*c +      hi*8, ka1, 0);
            load8_f16(kb + (16 + ql)*SKF + 32*c + 16 + hi*8, ka1, 8);
            acc0 = __builtin_amdgcn_wmma_f32_16x16x32_f16(false, ka0, false, qb[c],
                                                          (short)0, acc0, false, false);
            acc1 = __builtin_amdgcn_wmma_f32_16x16x32_f16(false, ka1, false, qb[c],
                                                          (short)0, acc1, false, false);
        }

        // ---- mask + scale: this lane holds keys {8*hi+j} and {16+8*hi+j}, j=0..7 ----
        const int4* mp = (const int4*)(mrow + kt);
        int4 m0 = mp[    hi*2], m1 = mp[    hi*2 + 1];  // keys 8*hi .. 8*hi+7
        int4 m2 = mp[4 + hi*2], m3 = mp[4 + hi*2 + 1];  // keys 16+8*hi ..
        int mm[16] = { m0.x,m0.y,m0.z,m0.w, m1.x,m1.y,m1.z,m1.w,
                       m2.x,m2.y,m2.z,m2.w, m3.x,m3.y,m3.z,m3.w };
        float sc[16];
        #pragma unroll
        for (int j = 0; j < 8; ++j) sc[j]   = mm[j]   ? acc0[j] * 0.125f : 1e-6f;
        #pragma unroll
        for (int j = 0; j < 8; ++j) sc[8+j] = mm[8+j] ? acc1[j] * 0.125f : 1e-6f;

        // ---- online softmax (per query; merge the two lane-halves) ----
        float mloc = sc[0];
        #pragma unroll
        for (int j = 1; j < 16; ++j) mloc = fmaxf(mloc, sc[j]);
        mloc = fmaxf(mloc, __shfl_xor(mloc, 16, 32));
        float mnew = fmaxf(m_run, mloc);
        float rescale = __expf(m_run - mnew);   // 0 on first iteration (m_run = -inf)
        float p[16], lsum = 0.0f;
        #pragma unroll
        for (int j = 0; j < 16; ++j) { p[j] = __expf(sc[j] - mnew); lsum += p[j]; }
        lsum += __shfl_xor(lsum, 16, 32);
        l_run = l_run * rescale + lsum;
        m_run = mnew;

        // ---- P^T is already in B-operand layout: pack to f16 in-register ----
        v16h pb;
        #pragma unroll
        for (int j = 0; j < 16; ++j) pb[j] = (_Float16)p[j];

        // ---- O^T += V^T x P^T : 4 d-chunks, K=32 keys each ----
        #pragma unroll
        for (int c = 0; c < 4; ++c) {
            v16h va;   // A operand: M = d row per lane, K = key swizzled
            *(v8h*)&va    = *(const v8h*)&ldsV[(c*16 + ql) * SVST +      hi*8];
            *((v8h*)&va+1)= *(const v8h*)&ldsV[(c*16 + ql) * SVST + 16 + hi*8];
            #pragma unroll
            for (int j = 0; j < 8; ++j) o[c][j] *= rescale;
            o[c] = __builtin_amdgcn_wmma_f32_16x16x32_f16(false, va, false, pb,
                                                          (short)0, o[c], false, false);
        }
    }

    // ---- epilogue: divide by softmax sum, store O[bh][q][d] ----
    float inv = 1.0f / l_run;
    float* orow = O + ((size_t)bh * SEQ + qg) * DK;
    #pragma unroll
    for (int c = 0; c < 4; ++c) {
        int d0 = c*16 + hi*8;   // this lane-half owns d = d0 .. d0+7
        float4 lo = { o[c][0]*inv, o[c][1]*inv, o[c][2]*inv, o[c][3]*inv };
        float4 hv = { o[c][4]*inv, o[c][5]*inv, o[c][6]*inv, o[c][7]*inv };
        ((float4*)(orow + d0    ))[0] = lo;
        ((float4*)(orow + d0 + 4))[0] = hv;
    }
}

extern "C" void kernel_launch(void* const* d_in, const int* in_sizes, int n_in,
                              void* d_out, int out_size, void* d_ws, size_t ws_size,
                              hipStream_t stream) {
    const float* Q   = (const float*)d_in[0];
    const float* K   = (const float*)d_in[1];
    const float* V   = (const float*)d_in[2];
    const int*   MSK = (const int*)d_in[3];
    float*       O   = (float*)d_out;

    dim3 grid(SEQ / BM, BHN);   // (16, 64): 64 queries per block, one bh per block.y
    attn_fwd_kernel<<<grid, 128, 0, stream>>>(Q, K, V, MSK, O);
}
